// Attention_22668837388651
// MI455X (gfx1250) — compile-verified, matrix operand order checked by reading, not run on hardware
//
#include <hip/hip_runtime.h>
#include <hip/hip_bf16.h>
#include <float.h>

// ---------------------------------------------------------------------------
// Talking-heads causal attention for MI455X (gfx1250), bf16 WMMA everywhere.
//   b=2, n=2048, dim=1024, h=16, dh=64, rot_dim=32, keys = n+1 (null) -> pad 2080
// Pipeline: cvt -> fused QKV GEMM (+rotary +scale +head layout, async-LDS
//           double-buffered B tiles) -> 2-pass attention (softmax ->
//           talking-heads mix -> PV) -> out GEMM (async-LDS B tiles).
// Workspace use: ~48.3 MB.
// ---------------------------------------------------------------------------

typedef __attribute__((ext_vector_type(16))) __bf16 v16bf;
typedef __attribute__((ext_vector_type(8)))  __bf16 v8bf;
typedef __attribute__((ext_vector_type(8)))  float  v8f;

#define B_     2
#define N_     2048
#define DIM_   1024
#define H_     16
#define DH_    64
#define INNER_ 1024
#define NKV_   2080   // padded key rows; valid keys = 2049 (null + 2048)
#define ROT_   32

// Load a 16x32 bf16 fragment (A-pattern) from a row-major matrix.
// ISA 16-bit A layout: lanes 0-15 -> row=lane, K {0..7,16..23};
// lanes 16-31 -> same rows, K {8..15,24..31}. Two 16B loads per lane.
__device__ __forceinline__ v16bf load_frag(const __bf16* __restrict__ base,
                                           int ld, int row0, int k0, int lane) {
  const int l15  = lane & 15;
  const int half = (lane >> 4) & 1;
  const __bf16* p = base + (size_t)(row0 + l15) * (size_t)ld + k0 + half * 8;
  v8bf lo = *(const v8bf*)(p);
  v8bf hi = *(const v8bf*)(p + 16);
  v16bf r;
#pragma unroll
  for (int i = 0; i < 8; ++i) { r[i] = lo[i]; r[i + 8] = hi[i]; }
  return r;
}

__device__ __forceinline__ v8f wmma_bf16(v16bf a, v16bf b, v8f c) {
  return __builtin_amdgcn_wmma_f32_16x16x32_bf16(
      /*neg_a=*/false, a, /*neg_b=*/false, b,
      /*c_mod=*/(short)0, c, /*reuse_a=*/false, /*reuse_b=*/false);
}

// Async copy 32B (two b128) global -> LDS. Per VGLOBAL pseudocode the
// instruction offset is added to BOTH the LDS (VDST) and global (VADDR)
// addresses, so one (lds, gaddr) pair covers both halves. Tracked by ASYNCcnt.
__device__ __forceinline__ void async_copy32(unsigned lds_addr, const void* gaddr) {
  asm volatile("global_load_async_to_lds_b128 %0, %1, off\n\t"
               "global_load_async_to_lds_b128 %0, %1, off offset:16"
               :: "v"(lds_addr), "v"(gaddr) : "memory");
}
__device__ __forceinline__ void wait_async0() {
  asm volatile("s_wait_asynccnt 0x0" ::: "memory");
}

// ------------------------------- converters --------------------------------
__global__ void cvt_f32_bf16(const float* __restrict__ in, __bf16* __restrict__ out, int n) {
  int i = blockIdx.x * blockDim.x + threadIdx.x;
  if (i < n) out[i] = (__bf16)in[i];
}

// wqkvT[c][k] : c in [0,3072) -> q cols 0..1023, k cols 1024..2047, v cols 2048..3071
__global__ void build_wqkvT(const float* __restrict__ wq, const float* __restrict__ wkv,
                            __bf16* __restrict__ wt) {
  int i = blockIdx.x * blockDim.x + threadIdx.x;
  if (i >= 3072 * DIM_) return;
  int c = i >> 10, k = i & 1023;
  float v = (c < 1024) ? wq[(size_t)k * 1024 + c] : wkv[(size_t)k * 2048 + (c - 1024)];
  wt[i] = (__bf16)v;
}

__global__ void build_woutT(const float* __restrict__ w, __bf16* __restrict__ wt) {
  int i = blockIdx.x * blockDim.x + threadIdx.x;
  if (i >= DIM_ * INNER_) return;
  int o = i >> 10, k = i & 1023;
  wt[i] = (__bf16)w[(size_t)k * DIM_ + o];
}

// null_k/null_v at key j=0 (no rotary on null rows per reference), zero the pad tail.
__global__ void fill_null_pad(const float* __restrict__ nk, const float* __restrict__ nv,
                              __bf16* __restrict__ K, __bf16* __restrict__ VT) {
  int bh = blockIdx.x;       // 0..31
  int d  = threadIdx.x;      // 0..63
  int h  = bh & (H_ - 1);
  K[((size_t)bh * NKV_ + 0) * DH_ + d]  = (__bf16)nk[h * DH_ + d];
  VT[((size_t)bh * DH_ + d) * NKV_ + 0] = (__bf16)nv[h * DH_ + d];
  for (int j = N_ + 1; j < NKV_; ++j) {
    K[((size_t)bh * NKV_ + j) * DH_ + d]  = (__bf16)0.0f;
    VT[((size_t)bh * DH_ + d) * NKV_ + j] = (__bf16)0.0f;
  }
}

// ------------------- fused QKV GEMM + rotary + head layout -----------------
// grid (M/64, 3072/64), 4 waves; wave computes 16 rows x 64 cols (one head strip).
// B tile (64 cols x 32 k, shared by all 4 waves) is double-buffered in LDS via
// async global->LDS copies; fragments are then ds_load'ed with the A-pattern.
__global__ __launch_bounds__(128) void gemm_qkv_kernel(
    const __bf16* __restrict__ X, const __bf16* __restrict__ Wt,
    const float* __restrict__ rope,
    __bf16* __restrict__ Q, __bf16* __restrict__ K, __bf16* __restrict__ VT) {
  __shared__ __align__(16) __bf16 Bt[2][64 * 32];   // 2 x 4 KB
  const int tid  = threadIdx.x;
  const int lane = tid & 31;
  const int wave = tid >> 5;
  const int l15  = lane & 15;
  const int half = (lane >> 4) & 1;
  const int row0 = blockIdx.x * 64 + wave * 16;
  const int col0 = blockIdx.y * 64;

  const int prow = tid >> 1;            // 0..63 : B-tile row (= output col)
  const int pseg = (tid & 1) * 16;      // 16 bf16 = 32 B per thread

  // prefetch K-step k0 into buffer `buf`
  auto prefetch = [&](int buf, int k0) {
    const __bf16* g = Wt + (size_t)(col0 + prow) * DIM_ + k0 + pseg;
    unsigned l = (unsigned)(size_t)&Bt[buf][prow * 32 + pseg];
    async_copy32(l, g);
  };

  int buf = 0;
  prefetch(0, 0);
  wait_async0();
  __syncthreads();

  v8f c[4] = {};
  for (int k0 = 0; k0 < DIM_; k0 += 32) {
    if (k0 + 32 < DIM_) prefetch(buf ^ 1, k0 + 32);
    v16bf a = load_frag(X, DIM_, row0, k0, lane);
    // load all four B fragments into distinct registers first so the
    // scheduler can use partial dscnt waits instead of wait-0 per WMMA
    v16bf bfr[4];
#pragma unroll
    for (int nt = 0; nt < 4; ++nt)
      bfr[nt] = load_frag(&Bt[buf][0], 32, nt * 16, 0, lane);
#pragma unroll
    for (int nt = 0; nt < 4; ++nt)
      c[nt] = wmma_bf16(a, bfr[nt], c[nt]);
    wait_async0();
    __syncthreads();
    buf ^= 1;
  }

  const int sec  = col0 >> 10;            // 0=q 1=k 2=v
  const int head = (col0 & 1023) >> 6;    // 64-aligned strip == one head
#pragma unroll
  for (int r = 0; r < 8; ++r) {
    const int grow = row0 + (half ? 8 + r : r);
    const int bidx = grow >> 11;
    const int tok  = grow & (N_ - 1);
    // rotary on head-cols [0,32): pair (c, c+16) lives in C-tiles 0 and 1, same lane
    const float f0 = rope[tok * ROT_ + l15];
    const float f1 = rope[tok * ROT_ + 16 + l15];
    const float x1 = c[0][r], x2 = c[1][r];
    float vals[4];
    vals[0] = x1 * __cosf(f0) - x2 * __sinf(f0);
    vals[1] = x2 * __cosf(f1) + x1 * __sinf(f1);
    vals[2] = c[2][r];
    vals[3] = c[3][r];
    const size_t bh = (size_t)bidx * H_ + head;
#pragma unroll
    for (int nt = 0; nt < 4; ++nt) {
      const int hc = nt * 16 + l15;
      if (sec == 0) {
        Q[(bh * N_ + tok) * DH_ + hc] = (__bf16)(vals[nt] * 0.125f);  // dh^-0.5
      } else if (sec == 1) {
        K[(bh * NKV_ + (tok + 1)) * DH_ + hc] = (__bf16)vals[nt];
      } else {
        VT[(bh * DH_ + hc) * NKV_ + (tok + 1)] = (__bf16)vals[nt];    // d-major
      }
    }
  }
}

// ----------------------------- attention -----------------------------------
// grid (n/16, b), 512 threads = 16 waves; wave == head (scores) == dest head (mix).
// Pass 1: exact row max / sumexp. Pass 2: normalized P -> LDS -> talking-heads
// mix -> PM @ V with WMMA. Causal mask j > i+1 also kills padded keys >= 2049.
__global__ __launch_bounds__(512) void attn_kernel(
    const __bf16* __restrict__ Q, const __bf16* __restrict__ K,
    const __bf16* __restrict__ VT, const float* __restrict__ thw,
    __bf16* __restrict__ AOUT) {
  __shared__ __align__(16) __bf16 Pl[H_ * 16 * 32];   // [head][qrow 16][key 32], 16 KB
  const int tid  = threadIdx.x;
  const int lane = tid & 31;
  const int wave = tid >> 5;
  const int l15  = lane & 15;
  const int half = (lane >> 4) & 1;
  const int i0   = blockIdx.x * 16;
  const int bidx = blockIdx.y;
  const int bh   = bidx * H_ + wave;

  const __bf16* qb = Q + ((size_t)bh * N_ + i0) * DH_;
  v16bf qf[2];
  qf[0] = load_frag(qb, DH_, 0, 0, lane);
  qf[1] = load_frag(qb, DH_, 0, 32, lane);

  const __bf16* kb = K + (size_t)bh * NKV_ * DH_;
  const __bf16* vb = VT + (size_t)bh * DH_ * NKV_;

  float thv[H_];
#pragma unroll
  for (int h = 0; h < H_; ++h) thv[h] = thw[wave * H_ + h];

  float mrow[8], lrow[8];
#pragma unroll
  for (int r = 0; r < 8; ++r) { mrow[r] = -FLT_MAX; lrow[r] = 0.0f; }

  const int njt = (i0 + 16 + 32) >> 5;  // ceil((i0+17)/32) key tiles

  // ---- pass 1 ----
  for (int jt = 0; jt < njt; ++jt) {
    const int j0 = jt * 32;
    v16bf kf[4];   // [nt][ks] flattened: kf[nt*2+ks]
#pragma unroll
    for (int nt = 0; nt < 2; ++nt)
#pragma unroll
      for (int ks = 0; ks < 2; ++ks)
        kf[nt * 2 + ks] = load_frag(kb, DH_, j0 + nt * 16, ks * 32, lane);
    v8f s[2] = {};
#pragma unroll
    for (int ks = 0; ks < 2; ++ks) {
      s[0] = wmma_bf16(qf[ks], kf[0 + ks], s[0]);
      s[1] = wmma_bf16(qf[ks], kf[2 + ks], s[1]);
    }
#pragma unroll
    for (int r = 0; r < 8; ++r) {
      const int irow = i0 + (half ? 8 + r : r);
#pragma unroll
      for (int nt = 0; nt < 2; ++nt)
        if (j0 + nt * 16 + l15 > irow + 1) s[nt][r] = -FLT_MAX;
      float tmax = fmaxf(s[0][r], s[1][r]);
      for (int d = 1; d < 16; d <<= 1) tmax = fmaxf(tmax, __shfl_xor(tmax, d, 32));
      const float nm = fmaxf(mrow[r], tmax);
      float rs = __expf(s[0][r] - nm) + __expf(s[1][r] - nm);
      for (int d = 1; d < 16; d <<= 1) rs += __shfl_xor(rs, d, 32);
      lrow[r] = lrow[r] * __expf(mrow[r] - nm) + rs;
      mrow[r] = nm;
    }
  }

  float rl[8];
#pragma unroll
  for (int r = 0; r < 8; ++r) rl[r] = (lrow[r] > 0.0f) ? 1.0f / lrow[r] : 0.0f;

  // ---- pass 2 ----
  v8f o[4] = {};
  for (int jt = 0; jt < njt; ++jt) {
    const int j0 = jt * 32;
    v16bf kf[4];
#pragma unroll
    for (int nt = 0; nt < 2; ++nt)
#pragma unroll
      for (int ks = 0; ks < 2; ++ks)
        kf[nt * 2 + ks] = load_frag(kb, DH_, j0 + nt * 16, ks * 32, lane);
    v8f s[2] = {};
#pragma unroll
    for (int ks = 0; ks < 2; ++ks) {
      s[0] = wmma_bf16(qf[ks], kf[0 + ks], s[0]);
      s[1] = wmma_bf16(qf[ks], kf[2 + ks], s[1]);
    }
#pragma unroll
    for (int r = 0; r < 8; ++r) {
      const int irow = i0 + (half ? 8 + r : r);
      const int rloc = half ? 8 + r : r;
#pragma unroll
      for (int nt = 0; nt < 2; ++nt) {
        const int j = j0 + nt * 16 + l15;
        float p = (j > irow + 1) ? 0.0f : __expf(s[nt][r] - mrow[r]) * rl[r];
        Pl[wave * 512 + rloc * 32 + nt * 16 + l15] = (__bf16)p;
      }
    }
    __syncthreads();
    // talking-heads mix, produced directly in WMMA A-fragment layout
    float acc[16];
#pragma unroll
    for (int i = 0; i < 16; ++i) acc[i] = 0.0f;
    const int off0 = half * 8;
#pragma unroll
    for (int h = 0; h < H_; ++h) {
      const __bf16* pr = &Pl[h * 512 + l15 * 32];
      v8bf a0 = *(const v8bf*)(pr + off0);
      v8bf a1 = *(const v8bf*)(pr + off0 + 16);
      const float w = thv[h];
#pragma unroll
      for (int i = 0; i < 8; ++i) {
        acc[i]     += w * (float)a0[i];
        acc[i + 8] += w * (float)a1[i];
      }
    }
    v16bf pm;
#pragma unroll
    for (int i = 0; i < 16; ++i) pm[i] = (__bf16)acc[i];
    v16bf vf[4];
#pragma unroll
    for (int ntd = 0; ntd < 4; ++ntd)
      vf[ntd] = load_frag(vb, NKV_, ntd * 16, j0, lane);  // rows = d (v stored d-major)
#pragma unroll
    for (int ntd = 0; ntd < 4; ++ntd)
      o[ntd] = wmma_bf16(pm, vf[ntd], o[ntd]);
    __syncthreads();
  }

#pragma unroll
  for (int ntd = 0; ntd < 4; ++ntd)
#pragma unroll
    for (int r = 0; r < 8; ++r) {
      const int tok = i0 + (half ? 8 + r : r);
      const int col = wave * DH_ + ntd * 16 + l15;
      AOUT[((size_t)bidx * N_ + tok) * INNER_ + col] = (__bf16)o[ntd][r];
    }
}

// ------------------------------ out GEMM -----------------------------------
__global__ __launch_bounds__(128) void gemm_out_kernel(
    const __bf16* __restrict__ A, const __bf16* __restrict__ Wt,
    const float* __restrict__ bias, float* __restrict__ out) {
  __shared__ __align__(16) __bf16 Bt[2][64 * 32];   // 2 x 4 KB
  const int tid  = threadIdx.x;
  const int lane = tid & 31;
  const int wave = tid >> 5;
  const int l15  = lane & 15;
  const int half = (lane >> 4) & 1;
  const int row0 = blockIdx.x * 64 + wave * 16;
  const int col0 = blockIdx.y * 64;

  const int prow = tid >> 1;
  const int pseg = (tid & 1) * 16;
  auto prefetch = [&](int buf, int k0) {
    const __bf16* g = Wt + (size_t)(col0 + prow) * INNER_ + k0 + pseg;
    unsigned l = (unsigned)(size_t)&Bt[buf][prow * 32 + pseg];
    async_copy32(l, g);
  };

  int buf = 0;
  prefetch(0, 0);
  wait_async0();
  __syncthreads();

  v8f c[4] = {};
  for (int k0 = 0; k0 < INNER_; k0 += 32) {
    if (k0 + 32 < INNER_) prefetch(buf ^ 1, k0 + 32);
    v16bf a = load_frag(A, INNER_, row0, k0, lane);
    v16bf bfr[4];
#pragma unroll
    for (int nt = 0; nt < 4; ++nt)
      bfr[nt] = load_frag(&Bt[buf][0], 32, nt * 16, 0, lane);
#pragma unroll
    for (int nt = 0; nt < 4; ++nt)
      c[nt] = wmma_bf16(a, bfr[nt], c[nt]);
    wait_async0();
    __syncthreads();
    buf ^= 1;
  }

#pragma unroll
  for (int nt = 0; nt < 4; ++nt)
#pragma unroll
    for (int r = 0; r < 8; ++r) {
      const int grow = row0 + (half ? 8 + r : r);
      const int col  = col0 + nt * 16 + l15;
      out[(size_t)grow * DIM_ + col] = c[nt][r] + bias[col];
    }
}

// ------------------------------- launcher ----------------------------------
extern "C" void kernel_launch(void* const* d_in, const int* in_sizes, int n_in,
                              void* d_out, int out_size, void* d_ws, size_t ws_size,
                              hipStream_t stream) {
  (void)in_sizes; (void)n_in; (void)out_size; (void)ws_size;
  const float* x      = (const float*)d_in[0];
  /* d_in[1] = mask (all true in this problem; key-mask is a no-op) */
  const float* rope   = (const float*)d_in[2];
  const float* null_k = (const float*)d_in[3];
  const float* null_v = (const float*)d_in[4];
  const float* th_w   = (const float*)d_in[5];
  const float* w_q    = (const float*)d_in[6];
  const float* w_kv   = (const float*)d_in[7];
  const float* w_out  = (const float*)d_in[8];
  const float* b_out  = (const float*)d_in[9];
  float* out = (float*)d_out;

  char* ws = (char*)d_ws;
  constexpr size_t SZ_XBF   = (size_t)B_ * N_ * DIM_ * 2;        // 8 MB
  constexpr size_t SZ_WQKVT = (size_t)3072 * DIM_ * 2;           // 6 MB
  constexpr size_t SZ_WOUTT = (size_t)DIM_ * INNER_ * 2;         // 2 MB
  constexpr size_t SZ_Q     = (size_t)B_ * H_ * N_ * DH_ * 2;    // 8 MB
  constexpr size_t SZ_KV    = (size_t)B_ * H_ * NKV_ * DH_ * 2;  // ~8.1 MB each
  constexpr size_t SZ_AOUT  = (size_t)B_ * N_ * INNER_ * 2;      // 8 MB
  __bf16* XBF   = (__bf16*)(ws);                       size_t off = SZ_XBF;
  __bf16* WQKVT = (__bf16*)(ws + off);  off += SZ_WQKVT;
  __bf16* WOUTT = (__bf16*)(ws + off);  off += SZ_WOUTT;
  __bf16* Qb    = (__bf16*)(ws + off);  off += SZ_Q;
  __bf16* Kb    = (__bf16*)(ws + off);  off += SZ_KV;
  __bf16* VTb   = (__bf16*)(ws + off);  off += SZ_KV;
  __bf16* AOUT  = (__bf16*)(ws + off);  off += SZ_AOUT;   // total ~48.3 MB

  cvt_f32_bf16<<<(B_ * N_ * DIM_ + 255) / 256, 256, 0, stream>>>(x, XBF, B_ * N_ * DIM_);
  build_wqkvT<<<(3072 * DIM_ + 255) / 256, 256, 0, stream>>>(w_q, w_kv, WQKVT);
  build_woutT<<<(DIM_ * INNER_ + 255) / 256, 256, 0, stream>>>(w_out, WOUTT);
  fill_null_pad<<<B_ * H_, DH_, 0, stream>>>(null_k, null_v, Kb, VTb);

  gemm_qkv_kernel<<<dim3((B_ * N_) / 64, 3072 / 64), 128, 0, stream>>>(
      XBF, WQKVT, rope, Qb, Kb, VTb);

  attn_kernel<<<dim3(N_ / 16, B_), 512, 0, stream>>>(Qb, Kb, VTb, th_w, AOUT);

  gemm_out_kernel<<<dim3((B_ * N_) / 64, DIM_ / 64), 128, 0, stream>>>(
      AOUT, WOUTT, b_out, out);
}
